// Net_61546881352193
// MI455X (gfx1250) — matrix operand order repaired and checked by reading, not compiled, not run on hardware
//
#include <hip/hip_runtime.h>
#include <hip/hip_bf16.h>

#define NN 128
#define TT 512
#define KK 256
#define VV 50257
#define NVT ((VV + 15) / 16)   /* 3142 v-tiles of 16 */
#define NBLK1 128
#define SMEM3 42112

typedef __bf16 bf16_t;
typedef bf16_t v16bf __attribute__((ext_vector_type(16)));
typedef float  v8f   __attribute__((ext_vector_type(8)));
typedef unsigned int   u32;
typedef u32 u32x4 __attribute__((ext_vector_type(4)));

__device__ __forceinline__ bf16_t f2bf(float f){ return (bf16_t)f; }

__device__ __forceinline__ void lse_acc(float& m, float& s, float x){
  if (x > m){ s = s * __expf(m - x) + 1.0f; m = x; }
  else       s += __expf(x - m);
}

// ---------------------------------------------------------------------------
// Kernel 1: logitsT[v][j] = sum_k Wcv[v][k] * E[j][k]   (bf16 WMMA, f32 acc)
// plus per-block online LSE partials over v for each j.
// B tiles (E^T) are vt-loop invariant: preload all 16 into registers.
// ---------------------------------------------------------------------------
__global__ void __launch_bounds__(256, 1)
k1_logits(const float* __restrict__ E,
          const float* __restrict__ Wcv,
          float* __restrict__ logitsT,
          float* __restrict__ maxpart,
          float* __restrict__ sumpart)
{
  extern __shared__ char smem1[];
  bf16_t* Et = (bf16_t*)smem1;                 // Et[j][k] = E[j][k], 128 KB
  int tid = threadIdx.x;
  int lane = tid & 31, wv = tid >> 5;
  int hi = lane >> 4, lq = lane & 15;

  for (int idx = tid; idx < KK*KK; idx += 256){
    int j = idx >> 8, k = idx & 255;
    Et[j*KK + k] = f2bf(E[idx]);
  }
  __syncthreads();

  // preload all B tiles for this wave's 32 columns into registers (128 VGPRs)
  v16bf breg[16];
  #pragma unroll
  for (int kb = 0; kb < 8; kb++){
    union { u32x4 q[2]; v16bf v; } ub0, ub1;
    const u32x4* bp0 = (const u32x4*)(Et + (wv*32 + lq)*KK + kb*32 + hi*8);
    ub0.q[0] = bp0[0]; ub0.q[1] = bp0[2];
    const u32x4* bp1 = (const u32x4*)(Et + (wv*32 + 16 + lq)*KK + kb*32 + hi*8);
    ub1.q[0] = bp1[0]; ub1.q[1] = bp1[2];
    breg[2*kb]   = ub0.v;
    breg[2*kb+1] = ub1.v;
  }

  int jc0 = wv*32 + lq;
  int jc1 = jc0 + 16;
  float m0 = -__builtin_inff(), s0 = 0.f;
  float m1 = -__builtin_inff(), s1 = 0.f;

  for (int vt = blockIdx.x; vt < NVT; vt += gridDim.x){
    v8f acc0 = {0,0,0,0,0,0,0,0};
    v8f acc1 = {0,0,0,0,0,0,0,0};
    int vA = vt*16 + lq;
    if (vA >= VV) vA = VV - 1;                 // clamp: rows >= VV are discarded below
    const float* arow = Wcv + (size_t)vA*KK;
    #pragma unroll
    for (int kb = 0; kb < 8; kb++){
      union { bf16_t h[16]; v16bf v; } ua;
      const float* p = arow + kb*32 + hi*8;
      #pragma unroll
      for (int i = 0; i < 8; i++){ ua.h[i] = f2bf(p[i]); ua.h[8+i] = f2bf(p[16+i]); }
      acc0 = __builtin_amdgcn_wmma_f32_16x16x32_bf16(false, ua.v, false, breg[2*kb],   (short)0, acc0, false, false);
      acc1 = __builtin_amdgcn_wmma_f32_16x16x32_bf16(false, ua.v, false, breg[2*kb+1], (short)0, acc1, false, false);
    }
    #pragma unroll
    for (int g = 0; g < 8; g++){
      int vrow = vt*16 + hi*8 + g;
      if (vrow < VV){
        float x0 = acc0[g], x1 = acc1[g];
        logitsT[(size_t)vrow*KK + jc0] = x0;
        logitsT[(size_t)vrow*KK + jc1] = x1;
        lse_acc(m0, s0, x0);
        lse_acc(m1, s1, x1);
      }
    }
  }
  // merge lane l <-> l^16 (same columns, complementary rows)
  {
    float om = __shfl_xor(m0, 16, 32), os = __shfl_xor(s0, 16, 32);
    float M = fmaxf(m0, om);
    float S = (M > -__builtin_inff()) ? s0*__expf(m0-M) + os*__expf(om-M) : 0.f;
    m0 = M; s0 = S;
    om = __shfl_xor(m1, 16, 32); os = __shfl_xor(s1, 16, 32);
    M = fmaxf(m1, om);
    S = (M > -__builtin_inff()) ? s1*__expf(m1-M) + os*__expf(om-M) : 0.f;
    m1 = M; s1 = S;
  }
  size_t base = (size_t)blockIdx.x * KK;
  if (lane < 16){ maxpart[base + jc0] = m0; sumpart[base + jc0] = s0; }
  else          { maxpart[base + jc1] = m1; sumpart[base + jc1] = s1; }
}

// ---------------------------------------------------------------------------
// Kernel 1b: reduce LSE partials -> lse[j]
// ---------------------------------------------------------------------------
__global__ void k1b_lse(const float* __restrict__ maxpart,
                        const float* __restrict__ sumpart,
                        float* __restrict__ lse)
{
  int j = threadIdx.x;
  float m = -__builtin_inff(), s = 0.f;
  for (int c = 0; c < NBLK1; c++){
    float cm = maxpart[(size_t)c*KK + j];
    float cs = sumpart[(size_t)c*KK + j];
    float M = fmaxf(m, cm);
    s = s*__expf(m - M) + cs*__expf(cm - M);
    m = M;
  }
  lse[j] = m + __logf(s);
}

// ---------------------------------------------------------------------------
// Kernel 2: Ttilde[i][j] = softmax over j of W_trans.reshape(K,K)   (bf16)
// ---------------------------------------------------------------------------
__global__ void k2_trans(const float* __restrict__ Wt, bf16_t* __restrict__ Tt)
{
  __shared__ float red[16];
  int i = blockIdx.x, tid = threadIdx.x, lane = tid & 31, wv = tid >> 5;
  float x = Wt[i*KK + tid];
  float m = x;
  #pragma unroll
  for (int off = 16; off >= 1; off >>= 1) m = fmaxf(m, __shfl_xor(m, off, 32));
  if (lane == 0) red[wv] = m;
  __syncthreads();
  float mf = red[0];
  #pragma unroll
  for (int k = 1; k < 8; k++) mf = fmaxf(mf, red[k]);
  float e = __expf(x - mf);
  float s = e;
  #pragma unroll
  for (int off = 16; off >= 1; off >>= 1) s += __shfl_xor(s, off, 32);
  if (lane == 0) red[8 + wv] = s;
  __syncthreads();
  float sf = 0.f;
  #pragma unroll
  for (int k = 0; k < 8; k++) sf += red[8 + k];
  Tt[tid*KK + i] = f2bf(e / sf);
}

// ---------------------------------------------------------------------------
// Kernel 2b: a0lin[k] = softmax(W_start[k] + b_start[k])   (f32)
// ---------------------------------------------------------------------------
__global__ void k2b_a0(const float* __restrict__ Ws, const float* __restrict__ bs,
                       float* __restrict__ a0)
{
  __shared__ float red[16];
  int tid = threadIdx.x, lane = tid & 31, wv = tid >> 5;
  float x = Ws[tid] + bs[tid];
  float m = x;
  #pragma unroll
  for (int off = 16; off >= 1; off >>= 1) m = fmaxf(m, __shfl_xor(m, off, 32));
  if (lane == 0) red[wv] = m;
  __syncthreads();
  float mf = red[0];
  #pragma unroll
  for (int k = 1; k < 8; k++) mf = fmaxf(mf, red[k]);
  float e = __expf(x - mf);
  float s = e;
  #pragma unroll
  for (int off = 16; off >= 1; off >>= 1) s += __shfl_xor(s, off, 32);
  if (lane == 0) red[8 + wv] = s;
  __syncthreads();
  float sf = 0.f;
  #pragma unroll
  for (int k = 0; k < 8; k++) sf += red[8 + k];
  a0[tid] = e / sf;
}

// ---------------------------------------------------------------------------
// Kernel 3: the forward scan. 8 WGs x 16 rows. Ttilde B tiles live in
// REGISTERS for the whole 511-step loop (loaded from global once).
// Per step: alphaLin @ Ttilde via WMMA, * emission prob, row-max renorm.
// ---------------------------------------------------------------------------
__global__ void __launch_bounds__(256, 1)
k3_scan(const int* __restrict__ w,
        const float* __restrict__ logitsT,
        const float* __restrict__ lse,
        const bf16_t* __restrict__ Ttg,
        const float* __restrict__ a0,
        float* __restrict__ partZ)
{
  extern __shared__ char smem[];
  bf16_t* al  = (bf16_t*)(smem);             //   8192 B : alphaLin bf16 [16][256]
  float* pls  = (float*)(smem + 8192);       //  16384 B : emission probs f32 [16][256]
  float* cur  = (float*)(smem + 24576);      //  16384 B : cur f32 [16][256]
  float* lsel = (float*)(smem + 40960);      //   1024 B : lse f32 [256]
  float* cl   = (float*)(smem + 41984);      //     64 B : log-scale c [16]
  float* mxl  = (float*)(smem + 42048);      //     64 B : last row max [16]

  int tid = threadIdx.x, lane = tid & 31, wv = tid >> 5;
  int hi = lane >> 4, lq = lane & 15;
  int r0 = blockIdx.x * 16;
  int jb0 = wv*32, jb1 = wv*32 + 16;

  // preload this wave's 32 Ttilde columns: all 16 B tiles -> 128 VGPRs
  v16bf breg[16];
  #pragma unroll
  for (int kb = 0; kb < 8; kb++){
    union { u32x4 q[2]; v16bf v; } ub0, ub1;
    const u32x4* bp0 = (const u32x4*)(Ttg + (jb0 + lq)*KK + kb*32 + hi*8);
    ub0.q[0] = bp0[0]; ub0.q[1] = bp0[2];
    const u32x4* bp1 = (const u32x4*)(Ttg + (jb1 + lq)*KK + kb*32 + hi*8);
    ub1.q[0] = bp1[0]; ub1.q[1] = bp1[2];
    breg[2*kb]   = ub0.v;
    breg[2*kb+1] = ub1.v;
  }

  for (int i = tid; i < 16*KK; i += 256) al[i] = f2bf(a0[i & 255]);
  lsel[tid] = lse[tid];
  if (tid < 16){ cl[tid] = 0.f; mxl[tid] = 1.f; }
  __syncthreads();

  for (int t = 1; t < TT; t++){
    {
      int m = tid >> 4, j0 = (tid & 15) * 16;
      int v = w[(r0 + m)*TT + t];                      // same addr for 16 threads -> broadcast
      const float* lrow = logitsT + (size_t)v*KK;
      #pragma unroll
      for (int jj = 0; jj < 16; jj++){
        int j = j0 + jj;
        pls[m*KK + j] = __expf(lrow[j] - lsel[j]);     // emission prob
      }
      if ((tid & 15) == 0 && (t + 1) < TT){            // prefetch next step's row
        int vn = w[(r0 + m)*TT + t + 1];
        __builtin_prefetch(logitsT + (size_t)vn*KK, 0, 1);
      }
    }
    __syncthreads();                                   // B: pls ready, al stable
    v8f acc0 = {0,0,0,0,0,0,0,0};
    v8f acc1 = {0,0,0,0,0,0,0,0};
    #pragma unroll
    for (int kb = 0; kb < 8; kb++){
      union { u32x4 q[2]; v16bf v; } ua;
      const u32x4* ap = (const u32x4*)(al + lq*KK + kb*32 + hi*8);
      ua.q[0] = ap[0];                                 // K = kb*32 + hi*8 .. +8
      ua.q[1] = ap[2];                                 // K = kb*32+16+hi*8 .. +8
      acc0 = __builtin_amdgcn_wmma_f32_16x16x32_bf16(false, ua.v, false, breg[2*kb],   (short)0, acc0, false, false);
      acc1 = __builtin_amdgcn_wmma_f32_16x16x32_bf16(false, ua.v, false, breg[2*kb+1], (short)0, acc1, false, false);
    }
    #pragma unroll
    for (int g = 0; g < 8; g++){
      int m = g + hi*8;
      int jA = jb0 + lq, jB = jb1 + lq;
      cur[m*KK + jA] = acc0[g] * pls[m*KK + jA];
      cur[m*KK + jB] = acc1[g] * pls[m*KK + jB];
    }
    __syncthreads();                                   // C: cur ready
    {
      int m = tid >> 4, q = tid & 15;
      float pm = -__builtin_inff();
      #pragma unroll
      for (int jj = q; jj < KK; jj += 16) pm = fmaxf(pm, cur[m*KK + jj]);
      #pragma unroll
      for (int off = 1; off < 16; off <<= 1) pm = fmaxf(pm, __shfl_xor(pm, off, 16));
      float inv = 1.0f / pm;
      #pragma unroll
      for (int jj = q; jj < KK; jj += 16) al[m*KK + jj] = f2bf(cur[m*KK + jj] * inv);
      if (q == 0){ cl[m] += __logf(pm); mxl[m] = pm; }
    }
    __syncthreads();                                   // D: al ready for next t
  }
  // logZ[m] = c_before_last + log(sum cur) = cl[m] - log(mxl[m]) + log(sum cur)
  {
    int m = tid >> 4, q = tid & 15;
    float s = 0.f;
    for (int jj = q; jj < KK; jj += 16) s += cur[m*KK + jj];
    #pragma unroll
    for (int off = 1; off < 16; off <<= 1) s += __shfl_xor(s, off, 16);
    if (q == 0) pls[m] = cl[m] - __logf(mxl[m]) + __logf(s);
  }
  __syncthreads();
  if (tid == 0){
    float acc = 0.f;
    for (int m = 0; m < 16; m++) acc += pls[m];
    partZ[blockIdx.x] = acc;
  }
}

// ---------------------------------------------------------------------------
// Kernel 4: out = -mean(logZ)
// ---------------------------------------------------------------------------
__global__ void k4_final(const float* __restrict__ partZ, float* __restrict__ out)
{
  if (threadIdx.x == 0){
    float s = 0.f;
    for (int i = 0; i < NN/16; i++) s += partZ[i];
    out[0] = -s / (float)NN;
  }
}

extern "C" void kernel_launch(void* const* d_in, const int* in_sizes, int n_in,
                              void* d_out, int out_size, void* d_ws, size_t ws_size,
                              hipStream_t stream)
{
  const int*   w      = (const int*)  d_in[0];
  const float* E      = (const float*)d_in[1];
  const float* Wcv    = (const float*)d_in[2];
  const float* Wstart = (const float*)d_in[3];
  const float* bstart = (const float*)d_in[4];
  const float* Wtr    = (const float*)d_in[5];

  char* ws = (char*)d_ws;
  size_t off = 0;
  float*  logitsT = (float*) (ws + off); off += (size_t)VV * KK * 4;   // 51.5 MB
  float*  maxpart = (float*) (ws + off); off += (size_t)NBLK1 * KK * 4;
  float*  sumpart = (float*) (ws + off); off += (size_t)NBLK1 * KK * 4;
  float*  lse     = (float*) (ws + off); off += KK * 4;
  bf16_t* Tt      = (bf16_t*)(ws + off); off += (size_t)KK * KK * 2;
  float*  a0      = (float*) (ws + off); off += KK * 4;
  float*  partZ   = (float*) (ws + off); off += 64;

  (void)hipFuncSetAttribute(reinterpret_cast<const void*>(k1_logits),
                            hipFuncAttributeMaxDynamicSharedMemorySize, 131072);

  k1_logits<<<NBLK1, 256, 131072, stream>>>(E, Wcv, logitsT, maxpart, sumpart);
  k1b_lse  <<<1, 256, 0, stream>>>(maxpart, sumpart, lse);
  k2_trans <<<KK, 256, 0, stream>>>(Wtr, Tt);
  k2b_a0   <<<1, 256, 0, stream>>>(Wstart, bstart, a0);
  k3_scan  <<<NN/16, 256, SMEM3, stream>>>(w, logitsT, lse, Tt, a0, partZ);
  k4_final <<<1, 64, 0, stream>>>(partZ, (float*)d_out);
}
